// SparseCHMM_20246475833996
// MI455X (gfx1250) — matrix operand order repaired and checked by reading, not compiled, not run on hardware
//
#include <hip/hip_runtime.h>
#include <math.h>

typedef __attribute__((ext_vector_type(16))) _Float16 v16h;
typedef __attribute__((ext_vector_type(8)))  float    v8f;
typedef __attribute__((ext_vector_type(4)))  float    v4f;

#define N_SRC   24
#define D_OBS   17
#define D_EMB   768
#define NN      289          // 17*17
#define MTOT    32768        // 64*512
#define LDA     776          // padded f16 row stride (768 + 8) to avoid bank conflicts

// ---------------------------------------------------------------------------
// Kernel 1: trans_logits = embs @ W_trans.T + b_trans  (M=32768, K=768, N=289)
// f16 hi/lo split WMMA (3x v_wmma_f32_16x16x32_f16 per K-step), fp32 accum.
// One 16-row M-tile per 128-thread (4-wave) block; each wave covers N-tiles
// {w, w+4, w+8, ...} of the 19 N-tiles (N padded 289 -> 304).
// ---------------------------------------------------------------------------
__global__ __launch_bounds__(128)
void trans_gemm_kernel(const float* __restrict__ embs,
                       const float* __restrict__ W,     // (289,768) row-major
                       const float* __restrict__ bias,  // (289)
                       float* __restrict__ out)         // logits (32768,289)
{
    __shared__ _Float16 Ah[16 * LDA];
    __shared__ _Float16 Al[16 * LDA];

    const int m0  = blockIdx.x * 16;
    const int tid = threadIdx.x;

    // Stage A tile (16 x 768) into LDS, pre-split into f16 hi/lo.
    for (int i = tid; i < 16 * D_EMB; i += 128) {
        int r = i / D_EMB, c = i - r * D_EMB;
        float x = embs[(size_t)(m0 + r) * D_EMB + c];
        _Float16 h = (_Float16)x;
        Ah[r * LDA + c] = h;
        Al[r * LDA + c] = (_Float16)(x - (float)h);
    }
    __syncthreads();

    const int wave = tid >> 5;
    const int lane = tid & 31;
    const int nrow = lane & 15;   // N-col within tile (B/C/D); M-row within tile (A)
    const int hi   = lane >> 4;   // half-wave select

    for (int t = wave; t < 19; t += 4) {
        const int n0 = t * 16;
        int nclamp = n0 + nrow; if (nclamp > NN - 1) nclamp = NN - 1;  // clamp B row
        const float* wrow = W + (size_t)nclamp * D_EMB;

        v8f acc = {};
        for (int kc = 0; kc < D_EMB; kc += 32) {
            // --- A fragments from LDS (lane: 16 contiguous K, matching B) ---
            const _Float16* arh = &Ah[nrow * LDA + kc + hi * 16];
            const _Float16* arl = &Al[nrow * LDA + kc + hi * 16];
            v16h a_h, a_l;
            #pragma unroll
            for (int i = 0; i < 8; ++i) {
                a_h[i]     = arh[i];
                a_h[8 + i] = arh[8 + i];
                a_l[i]     = arl[i];
                a_l[8 + i] = arl[8 + i];
            }
            // --- B fragments from global (lane: 16 contiguous K) ---
            const v4f* wb4 = (const v4f*)(wrow + kc + hi * 16);
            v16h b_h, b_l;
            #pragma unroll
            for (int q = 0; q < 4; ++q) {
                v4f x4 = wb4[q];
                #pragma unroll
                for (int e = 0; e < 4; ++e) {
                    float x = x4[e];
                    _Float16 h = (_Float16)x;
                    b_h[q * 4 + e] = h;
                    b_l[q * 4 + e] = (_Float16)(x - (float)h);
                }
            }
            // hi/lo split products, f32 accumulate (Al*Bl dropped, ~2^-22 rel)
            acc = __builtin_amdgcn_wmma_f32_16x16x32_f16(false, a_h, false, b_h,
                                                         (short)0, acc, false, false);
            acc = __builtin_amdgcn_wmma_f32_16x16x32_f16(false, a_l, false, b_h,
                                                         (short)0, acc, false, false);
            acc = __builtin_amdgcn_wmma_f32_16x16x32_f16(false, a_h, false, b_l,
                                                         (short)0, acc, false, false);
        }
        const int nc = n0 + nrow;
        if (nc < NN) {
            float bv = bias[nc];
            #pragma unroll
            for (int r = 0; r < 8; ++r) {
                int m = m0 + r + hi * 8;                 // C/D: M = r + 8*(lane>=16)
                out[(size_t)m * NN + nc] = acc[r] + bv;
            }
        }
    }
}

// ---------------------------------------------------------------------------
// Kernel 2: in-place softmax over rows of 17 (trans_probs slot holds logits)
// ---------------------------------------------------------------------------
__global__ __launch_bounds__(256)
void softmax17_kernel(float* __restrict__ data, int nrows)
{
    int row = blockIdx.x * 256 + threadIdx.x;
    if (row >= nrows) return;
    float* p = data + (size_t)row * 17;
    float v[17];
    float mx = -INFINITY;
    #pragma unroll
    for (int i = 0; i < 17; ++i) { v[i] = p[i]; mx = fmaxf(mx, v[i]); }
    float s = 0.f;
    #pragma unroll
    for (int i = 0; i < 17; ++i) { v[i] = expf(v[i] - mx); s += v[i]; }
    float inv = 1.f / s;
    #pragma unroll
    for (int i = 0; i < 17; ++i) p[i] = v[i] * inv;
}

// ---------------------------------------------------------------------------
// Kernel 3: per-batch emission pipeline. One block (256 thr) per batch.
// ---------------------------------------------------------------------------
__device__ __forceinline__ float sigm(float x) { return 1.f / (1.f + expf(-x)); }

__global__ __launch_bounds__(256)
void emiss_kernel(const float* __restrict__ embs,
                  const float* __restrict__ W2o,  const float* __restrict__ b2o,
                  const float* __restrict__ W2e,  const float* __restrict__ b2e,
                  const float* __restrict__ Wexp, const float* __restrict__ bexp,
                  const float* __restrict__ wxor_lut,     // (24,16,16)
                  float* __restrict__ out_emiss,          // (64,24,17,17)
                  float* __restrict__ out_o2o,            // (64,24)
                  float* __restrict__ out_l2l,            // (64,24,16)
                  float* __restrict__ out_swx)            // (64,24,16,16)
{
    __shared__ float x0[D_EMB];
    __shared__ float res[792];        // 24 o2o | 384 e2e | 384 exp (bias added)
    __shared__ float o2o[24];
    __shared__ float l2l[384];
    __shared__ float nond[384];
    __shared__ float oval[384];
    __shared__ float wsc[384];
    __shared__ float swx[6144];       // (24,16,16)

    const int b   = blockIdx.x;
    const int tid = threadIdx.x;
    const int wave = tid >> 5, lane = tid & 31;

    for (int i = tid; i < D_EMB; i += 256)
        x0[i] = embs[(size_t)b * 512 * D_EMB + i];       // embs[b, 0, :]
    __syncthreads();

    // 792 row dot-products, one wave per row round-robin
    for (int r = wave; r < 792; r += 8) {
        const float* Wp; float bv;
        if (r < 24)       { Wp = W2o  + (size_t)r * D_EMB;         bv = b2o[r]; }
        else if (r < 408) { Wp = W2e  + (size_t)(r - 24) * D_EMB;  bv = b2e[r - 24]; }
        else              { Wp = Wexp + (size_t)(r - 408) * D_EMB; bv = bexp[r - 408]; }
        float p = 0.f;
        for (int k = lane; k < D_EMB; k += 32) p = fmaf(Wp[k], x0[k], p);
        #pragma unroll
        for (int off = 16; off; off >>= 1) p += __shfl_xor(p, off, 32);
        if (lane == 0) res[r] = p + bv;
    }
    __syncthreads();

    if (tid < 24) o2o[tid] = sigm(res[tid]);
    if (tid < 16) {
        // softmax over the 24 sources for column c, then prob_scaling
        const int c = tid;
        float mx = -INFINITY;
        for (int s = 0; s < 24; ++s) mx = fmaxf(mx, res[24 + s * 16 + c]);
        float sum = 0.f, e[24];
        for (int s = 0; s < 24; ++s) { e[s] = expf(res[24 + s * 16 + c] - mx); sum += e[s]; }
        const float rr = 1.f / 24.f;
        const float bb = 1.f / ((1.f - rr) * (1.f - rr) * (1.f - rr)); // (1-r^e1)/(1-r)^e2, e1=1,e2=4
        for (int s = 0; s < 24; ++s) {
            float x = e[s] / sum;
            float t = 1.f - x;
            float t2 = t * t;
            l2l[s * 16 + c] = (x < rr) ? x : 1.f - bb * t2 * t2;
        }
    }
    __syncthreads();

    for (int i = tid; i < 384; i += 256) {
        float x = l2l[i];
        const float tp = 1.f / 17.f;                     // NONDIAG_R / d
        float xr = x / tp;
        float wf = (x < tp) ? tp * xr * xr : x;          // _wrong_frac, exp=2
        nond[i] = (1.f - x) * wf * (1.f / 15.f);         // /(d-2)
        oval[i] = (1.f - x) * (1.f - wf);
        wsc[i]  = sigm(res[408 + i]);
        out_l2l[(size_t)b * 384 + i] = x;
    }
    if (tid < 24) out_o2o[(size_t)b * 24 + tid] = o2o[tid];
    __syncthreads();

    // scaled_wxor[b,s,i,j] = wxor_scale[b,s,j] * wxor_lut[s,j,i]
    for (int idx = tid; idx < 6144; idx += 256) {
        int s = idx >> 8, rem = idx & 255;
        int i = rem >> 4, j = rem & 15;
        float v = wsc[s * 16 + j] * wxor_lut[s * 256 + j * 16 + i];
        swx[idx] = v;
        out_swx[(size_t)b * 6144 + idx] = v;
    }
    __syncthreads();

    // Dirichlet rows: 24 sources x 17 rows of 17
    for (int task = tid; task < 24 * 17; task += 256) {
        int s = task / 17, row = task % 17;
        float v[17];
        if (row == 0) {
            v[0] = o2o[s];
            float f = (1.f - o2o[s]) * (1.f / 16.f);
            #pragma unroll
            for (int j = 0; j < 16; ++j) v[1 + j] = f;
        } else {
            int i = row - 1;
            v[0] = oval[s * 16 + i];
            float dg = l2l[s * 16 + i], nd = nond[s * 16 + i];
            #pragma unroll
            for (int j = 0; j < 16; ++j)
                v[1 + j] = ((i == j) ? dg : nd) + swx[s * 256 + i * 16 + j];
        }
        float sum = 0.f;
        #pragma unroll
        for (int k = 0; k < 17; ++k) { v[k] = v[k] * 99.f + 1.f; sum += v[k]; } // *(100-1)+1
        float inv = 1.f / sum;
        float* op = out_emiss + (size_t)b * 24 * NN + (size_t)s * NN + (size_t)row * 17;
        #pragma unroll
        for (int k = 0; k < 17; ++k) op[k] = v[k] * inv;
    }
}

// ---------------------------------------------------------------------------
extern "C" void kernel_launch(void* const* d_in, const int* in_sizes, int n_in,
                              void* d_out, int out_size, void* d_ws, size_t ws_size,
                              hipStream_t stream) {
    (void)in_sizes; (void)n_in; (void)out_size; (void)d_ws; (void)ws_size;

    const float* embs   = (const float*)d_in[0];
    const float* Wtr    = (const float*)d_in[1];
    const float* btr    = (const float*)d_in[2];
    const float* W2o    = (const float*)d_in[3];
    const float* b2o    = (const float*)d_in[4];
    const float* W2e    = (const float*)d_in[5];
    const float* b2e    = (const float*)d_in[6];
    const float* Wexp   = (const float*)d_in[7];
    const float* bexp   = (const float*)d_in[8];
    const float* wlut   = (const float*)d_in[9];

    float* out = (float*)d_out;
    const size_t OFF_TRANS = 0;
    const size_t OFF_EMISS = (size_t)MTOT * NN;                 //  9,469,952
    const size_t OFF_O2O   = OFF_EMISS + (size_t)64 * 24 * NN;  //  9,913,856
    const size_t OFF_L2L   = OFF_O2O   + (size_t)64 * 24;       //  9,915,392
    const size_t OFF_SWX   = OFF_L2L   + (size_t)64 * 384;      //  9,939,968

    // 1) GEMM -> logits into trans_probs slot
    trans_gemm_kernel<<<MTOT / 16, 128, 0, stream>>>(embs, Wtr, btr, out + OFF_TRANS);

    // 2) in-place softmax over rows of 17
    const int nrows = MTOT * 17;                                // 557,056
    softmax17_kernel<<<(nrows + 255) / 256, 256, 0, stream>>>(out + OFF_TRANS, nrows);

    // 3) per-batch emission pipeline
    emiss_kernel<<<64, 256, 0, stream>>>(embs, W2o, b2o, W2e, b2e, Wexp, bexp, wlut,
                                         out + OFF_EMISS, out + OFF_O2O,
                                         out + OFF_L2L, out + OFF_SWX);
}
